// HelenaNet_49374944034997
// MI455X (gfx1250) — matrix operand order repaired and checked by reading, not compiled, not run on hardware
//
#include <hip/hip_runtime.h>
#include <hip/hip_bf16.h>
#include <math.h>

// ---------------------------------------------------------------------------
// HelenaNet forward for gfx1250 (MI455X).  All heavy GEMMs run through
// v_wmma_f32_16x16x32_bf16 (bf16 inputs converted on the fly, fp32 acc).
// GEMM: 128x64 block tile, 8 waves, K staged 64-deep, double-buffered LDS,
// branch-free vectorized staging (b128 loads -> cvt_pk_bf16 -> b64 stores),
// fragment loads grouped ahead of 4x back-to-back WMMA issues.
// ---------------------------------------------------------------------------

typedef __attribute__((ext_vector_type(16))) __bf16 v16bf;
typedef __attribute__((ext_vector_type(4)))  __bf16 v4bf;
typedef __attribute__((ext_vector_type(8)))  float  v8f;

#define LDS_PITCH 72   // bf16 elems per row slot (64 data + 8 pad); 144B rows

__device__ __forceinline__ v4bf pack4(float4 f) {
    v4bf r;
    r[0] = (__bf16)f.x; r[1] = (__bf16)f.y; r[2] = (__bf16)f.z; r[3] = (__bf16)f.w;
    return r;
}

// ---------------------------------------------------------------------------
// C[M,N] (+)= A[M,K] @ W[N,K]^T ;  A:[M,K] f32, W:[N,K] f32 (torch Linear
// layout), C:[M,N] f32.  Requires K % 64 == 0 (true for every call site:
// K is 512 or 1024).  M/N handled by clamped loads + guarded stores.
// Block: 256 threads (8 waves). Tile: 128(M) x 64(N), K staged 64 deep.
// Wave wv owns M rows [wv*16, wv*16+16) and all 64 N cols (4 accumulators).
// ---------------------------------------------------------------------------
__global__ __launch_bounds__(256) void gemm_wmma_bf16(
    const float* __restrict__ A, int lda,
    const float* __restrict__ W, int ldw,
    float* __restrict__ C, int ldc,
    int M, int N, int K, int accumulate)
{
    // double-buffered tiles: A 128x64 bf16, B(W rows) 64x64 bf16
    __shared__ __bf16 sA[2][128 * LDS_PITCH];
    __shared__ __bf16 sB[2][64 * LDS_PITCH];

    const int tid  = threadIdx.x;
    const int m0   = blockIdx.y * 128;
    const int n0   = blockIdx.x * 64;
    const int wv   = tid >> 5;
    const int lane = tid & 31;

    v8f acc[4] = {v8f{}, v8f{}, v8f{}, v8f{}};

    if (accumulate) {
        #pragma unroll
        for (int n = 0; n < 4; ++n) {
            #pragma unroll
            for (int v = 0; v < 8; ++v) {
                int gr = m0 + wv * 16 + ((lane < 16) ? v : v + 8);
                int gc = n0 + n * 16 + (lane & 15);
                if (gr < M && gc < N) acc[n][v] = C[(size_t)gr * ldc + gc];
            }
        }
    }

    const int nk = K >> 6;            // number of 64-deep K stages

    // registers holding the staged (next) tile:
    //   A: 128x64 = 2048 float4 chunks / 256 thr = 8 ;  B: 64x64 = 4
    float4 ra[8], rb[4];

    auto gload = [&](int k0) {
        #pragma unroll
        for (int p = 0; p < 8; ++p) {
            int chunk = tid + p * 256;
            int r  = chunk >> 4;              // 16 chunks per row
            int c4 = (chunk & 15) << 2;
            int gr = m0 + r; if (gr >= M) gr = M - 1;
            ra[p] = *(const float4*)(A + (size_t)gr * lda + k0 + c4);
        }
        #pragma unroll
        for (int p = 0; p < 4; ++p) {
            int chunk = tid + p * 256;
            int r  = chunk >> 4;
            int c4 = (chunk & 15) << 2;
            int gn = n0 + r; if (gn >= N) gn = N - 1;
            rb[p] = *(const float4*)(W + (size_t)gn * ldw + k0 + c4);
        }
    };
    auto stage = [&](int buf) {
        #pragma unroll
        for (int p = 0; p < 8; ++p) {
            int chunk = tid + p * 256;
            int r  = chunk >> 4;
            int c4 = (chunk & 15) << 2;
            *(v4bf*)&sA[buf][r * LDS_PITCH + c4] = pack4(ra[p]);
        }
        #pragma unroll
        for (int p = 0; p < 4; ++p) {
            int chunk = tid + p * 256;
            int r  = chunk >> 4;
            int c4 = (chunk & 15) << 2;
            *(v4bf*)&sB[buf][r * LDS_PITCH + c4] = pack4(rb[p]);
        }
    };

    // fragment addressing
    const int rowA  = wv * 16 + (lane & 15);
    const int khalf = (lane < 16) ? 0 : 16;   // A: K 0..15 / 16..31 across lane halves
    const int kbB   = (lane < 16) ? 0 : 16;   // B: K 0..15 / 16..31 across lane halves
    const int colB  = lane & 15;

    gload(0);
    stage(0);

    for (int kt = 0; kt < nk; ++kt) {
        __syncthreads();
        if (kt + 1 < nk) gload((kt + 1) << 6);   // overlap with WMMA below

        const int buf = kt & 1;
        #pragma unroll
        for (int sub = 0; sub < 2; ++sub) {      // two 32-deep K sub-steps
            const int kk = sub << 5;
            // fetch A frag and ALL FOUR B frags first, then issue 4 WMMAs
            v16bf a, b0, b1, b2, b3;
            #pragma unroll
            for (int i = 0; i < 8; ++i) {
                a[i]     = sA[buf][rowA * LDS_PITCH + kk + khalf + i];
                a[i + 8] = sA[buf][rowA * LDS_PITCH + kk + khalf + 8 + i];
            }
            #pragma unroll
            for (int i = 0; i < 16; ++i) {
                b0[i] = sB[buf][(0 * 16 + colB) * LDS_PITCH + kk + kbB + i];
                b1[i] = sB[buf][(1 * 16 + colB) * LDS_PITCH + kk + kbB + i];
                b2[i] = sB[buf][(2 * 16 + colB) * LDS_PITCH + kk + kbB + i];
                b3[i] = sB[buf][(3 * 16 + colB) * LDS_PITCH + kk + kbB + i];
            }
            acc[0] = __builtin_amdgcn_wmma_f32_16x16x32_bf16(
                         false, a, false, b0, (short)0, acc[0], false, false);
            acc[1] = __builtin_amdgcn_wmma_f32_16x16x32_bf16(
                         false, a, false, b1, (short)0, acc[1], false, false);
            acc[2] = __builtin_amdgcn_wmma_f32_16x16x32_bf16(
                         false, a, false, b2, (short)0, acc[2], false, false);
            acc[3] = __builtin_amdgcn_wmma_f32_16x16x32_bf16(
                         false, a, false, b3, (short)0, acc[3], false, false);
        }

        if (kt + 1 < nk) stage((kt + 1) & 1);    // other buffer: safe post-barrier
    }

    #pragma unroll
    for (int n = 0; n < 4; ++n) {
        #pragma unroll
        for (int v = 0; v < 8; ++v) {
            int gr = m0 + wv * 16 + ((lane < 16) ? v : v + 8);
            int gc = n0 + n * 16 + (lane & 15);
            if (gr < M && gc < N) C[(size_t)gr * ldc + gc] = acc[n][v];
        }
    }
}

// ---------------------------------------------------------------------------
// Elementwise / recurrence kernels (fp32 VALU)
// ---------------------------------------------------------------------------
__device__ __forceinline__ float siluf(float x) { return x / (1.0f + expf(-x)); }

__global__ void embed_kernel(const int* __restrict__ tok,
                             const float* __restrict__ emb,
                             const float* __restrict__ pos,
                             float* __restrict__ x, int n, int D)
{
    int i = blockIdx.x * blockDim.x + threadIdx.x;
    if (i >= n) return;
    int s = i / D, d = i % D;
    x[i] = emb[(size_t)tok[s] * D + d] + pos[i];
}

__global__ void rms_kernel(const float* __restrict__ in,
                           const float* __restrict__ w,
                           float* __restrict__ out, int D)
{
    __shared__ float red[256];
    int s = blockIdx.x;
    const float* row = in + (size_t)s * D;
    float acc = 0.0f;
    for (int d = threadIdx.x; d < D; d += 256) { float v = row[d]; acc += v * v; }
    red[threadIdx.x] = acc;
    __syncthreads();
    for (int st = 128; st > 0; st >>= 1) {
        if (threadIdx.x < st) red[threadIdx.x] += red[threadIdx.x + st];
        __syncthreads();
    }
    float scale = rsqrtf(red[0] / (float)D + 1e-6f);
    for (int d = threadIdx.x; d < D; d += 256)
        out[(size_t)s * D + d] = row[d] * scale * w[d];
}

// Depthwise causal conv (width 4) + SiLU; xb = first Di channels of xz.
__global__ void conv_silu_kernel(const float* __restrict__ xz,
                                 const float* __restrict__ cw,
                                 const float* __restrict__ cb,
                                 float* __restrict__ xc, int S, int Di)
{
    int i = blockIdx.x * blockDim.x + threadIdx.x;
    if (i >= S * Di) return;
    int s = i / Di, d = i % Di;
    float a = cb[d];
    #pragma unroll
    for (int j = 0; j < 4; ++j) {
        int sp = s - 3 + j;
        if (sp >= 0) a += cw[d * 4 + j] * xz[(size_t)sp * 2 * Di + d];
    }
    xc[i] = siluf(a);
}

__global__ void dt_kernel(const float* __restrict__ xdbl,
                          const float* __restrict__ dtw,
                          const float* __restrict__ dtb,
                          float* __restrict__ dt, int S, int Di)
{
    int i = blockIdx.x * blockDim.x + threadIdx.x;
    if (i >= S * Di) return;
    int s = i / Di, d = i % Di;
    float v = xdbl[(size_t)s * 33] * dtw[d] + dtb[d];
    dt[i] = (v > 20.0f) ? v : log1pf(expf(v));   // softplus
}

// Selective scan: per channel d, 16-wide state in registers, loop s.
__global__ void ssm_scan_kernel(const float* __restrict__ dt,
                                const float* __restrict__ xc,
                                const float* __restrict__ xdbl,
                                const float* __restrict__ A_log,
                                const float* __restrict__ Dsk,
                                const float* __restrict__ xz,
                                float* __restrict__ y, int S, int Di)
{
    int d = blockIdx.x * blockDim.x + threadIdx.x;
    if (d >= Di) return;
    float A[16], h[16];
    #pragma unroll
    for (int j = 0; j < 16; ++j) { A[j] = -expf(A_log[(size_t)d * 16 + j]); h[j] = 0.0f; }
    const float dskip = Dsk[d];
    for (int s = 0; s < S; ++s) {
        float dtv = dt[(size_t)s * Di + d];
        float xcv = xc[(size_t)s * Di + d];
        const float* Bs = xdbl + (size_t)s * 33 + 1;
        const float* Cs = Bs + 16;
        float accv = 0.0f;
        #pragma unroll
        for (int j = 0; j < 16; ++j) {
            h[j] = expf(dtv * A[j]) * h[j] + dtv * Bs[j] * xcv;
            accv += h[j] * Cs[j];
        }
        float z = xz[(size_t)s * 2 * Di + Di + d];
        y[(size_t)s * Di + d] = (accv + xcv * dskip) * siluf(z);
    }
}

// LIF membrane: m = LEAK*m + I; hard spike; subtract threshold.
__global__ void lif_scan_kernel(const float* __restrict__ cur,
                                float* __restrict__ spk, int S, int D)
{
    int d = blockIdx.x * blockDim.x + threadIdx.x;
    if (d >= D) return;
    float m = 0.0f;
    for (int s = 0; s < S; ++s) {
        m = 0.9f * m + cur[(size_t)s * D + d];
        float spike = (m >= 1.0f) ? 1.0f : 0.0f;
        spk[(size_t)s * D + d] = spike;
        m -= spike;   // spike * THRESH, THRESH = 1
    }
}

// Router: logits -> softmax over 8 -> top-2 mask -> combine weights.
__global__ void router_kernel(const float* __restrict__ xn,
                              const float* __restrict__ rw,
                              float* __restrict__ wts, int T, int D)
{
    int t = blockIdx.x * blockDim.x + threadIdx.x;
    if (t >= T) return;
    const float* xr = xn + (size_t)t * D;
    float lg[8];
    for (int e = 0; e < 8; ++e) {
        const float* we = rw + (size_t)e * D;
        float a = 0.0f;
        for (int d = 0; d < D; ++d) a += xr[d] * we[d];
        lg[e] = a;
    }
    float mx = lg[0];
    for (int e = 1; e < 8; ++e) mx = fmaxf(mx, lg[e]);
    float p[8], sum = 0.0f;
    for (int e = 0; e < 8; ++e) { p[e] = expf(lg[e] - mx); sum += p[e]; }
    for (int e = 0; e < 8; ++e) p[e] /= sum;
    int i1 = 0;
    for (int e = 1; e < 8; ++e) if (p[e] > p[i1]) i1 = e;
    int i2 = (i1 == 0) ? 1 : 0;
    for (int e = 0; e < 8; ++e) { if (e == i1) continue; if (p[e] > p[i2]) i2 = e; }
    for (int e = 0; e < 8; ++e)
        wts[(size_t)t * 8 + e] = (e == i1 || e == i2) ? p[e] : 0.0f;
}

// hbw = silu(gate) * up * route_weight  (in place into `he`)
__global__ void moe_mul_kernel(float* __restrict__ he,
                               const float* __restrict__ hu,
                               const float* __restrict__ wts,
                               int e, int T, int H)
{
    int i = blockIdx.x * blockDim.x + threadIdx.x;
    if (i >= T * H) return;
    int t = i / H;
    he[i] = siluf(he[i]) * hu[i] * wts[(size_t)t * 8 + e];
}

// ---------------------------------------------------------------------------
// Orchestration
// ---------------------------------------------------------------------------
extern "C" void kernel_launch(void* const* d_in, const int* in_sizes, int n_in,
                              void* d_out, int out_size, void* d_ws, size_t ws_size,
                              hipStream_t stream)
{
    (void)in_sizes; (void)n_in; (void)out_size; (void)ws_size;

    const int S = 1024, D = 512, Di = 1024, L = 4, E = 8, H = 1024, V = 32000, T = 1024;

    const int*   tok      = (const int*)  d_in[0];
    const float* emb      = (const float*)d_in[1];
    const float* pos      = (const float*)d_in[2];
    const float* ssm_norm = (const float*)d_in[3];
    const float* in_proj  = (const float*)d_in[4];
    const float* conv_w   = (const float*)d_in[5];
    const float* conv_b   = (const float*)d_in[6];
    const float* A_log    = (const float*)d_in[7];
    const float* Dsk      = (const float*)d_in[8];
    const float* x_proj   = (const float*)d_in[9];
    const float* dtw      = (const float*)d_in[10];
    const float* dtb      = (const float*)d_in[11];
    const float* out_proj = (const float*)d_in[12];
    const float* lif_norm = (const float*)d_in[13];
    const float* lif_in   = (const float*)d_in[14];
    const float* lif_out  = (const float*)d_in[15];
    const float* moe_norm = (const float*)d_in[16];
    const float* router   = (const float*)d_in[17];
    const float* gate     = (const float*)d_in[18];
    const float* up       = (const float*)d_in[19];
    const float* down     = (const float*)d_in[20];
    const float* norm_out = (const float*)d_in[21];
    const float* lm_head  = (const float*)d_in[22];
    float*       outp     = (float*)d_out;

    // Workspace layout (floats): ~38 MB, all written before read.
    float* ws = (float*)d_ws;
    size_t off = 0;
    float* x     = ws + off; off += (size_t)S * D;        // residual stream
    float* xn    = ws + off; off += (size_t)S * D;        // rms-normed
    float* xz    = ws + off; off += (size_t)S * 2 * Di;   // in_proj out
    float* xc    = ws + off; off += (size_t)S * Di;       // conv+silu out
    float* xdbl  = ws + off; off += (size_t)S * 33;       // x_proj out
    float* dtbuf = ws + off; off += (size_t)S * Di;       // softplus(dt)
    float* yb    = ws + off; off += (size_t)S * Di;       // scan out
    float* cur   = ws + off; off += (size_t)S * D;        // lif current
    float* spk   = ws + off; off += (size_t)S * D;        // lif spikes
    float* he    = ws + off; off += (size_t)T * H;        // expert gate / hbw
    float* hu    = ws + off; off += (size_t)T * H;        // expert up
    float* wts   = ws + off; off += (size_t)T * E;        // router weights

    auto gemm = [&](const float* A, int lda, const float* W, int ldw,
                    float* C, int ldc, int M, int N, int K, int acc) {
        dim3 grid((N + 63) / 64, (M + 127) / 128);
        gemm_wmma_bf16<<<grid, 256, 0, stream>>>(A, lda, W, ldw, C, ldc, M, N, K, acc);
    };

    embed_kernel<<<(S * D + 255) / 256, 256, 0, stream>>>(tok, emb, pos, x, S * D, D);

    for (int l = 0; l < L; ++l) {
        // ---- SSM layer ----
        rms_kernel<<<S, 256, 0, stream>>>(x, ssm_norm + (size_t)l * D, xn, D);
        gemm(xn, D, in_proj + (size_t)l * 2 * Di * D, D, xz, 2 * Di, S, 2 * Di, D, 0);
        conv_silu_kernel<<<(S * Di + 255) / 256, 256, 0, stream>>>(
            xz, conv_w + (size_t)l * Di * 4, conv_b + (size_t)l * Di, xc, S, Di);
        gemm(xc, Di, x_proj + (size_t)l * 33 * Di, Di, xdbl, 33, S, 33, Di, 0);
        dt_kernel<<<(S * Di + 255) / 256, 256, 0, stream>>>(
            xdbl, dtw + (size_t)l * Di, dtb + (size_t)l * Di, dtbuf, S, Di);
        ssm_scan_kernel<<<(Di + 255) / 256, 256, 0, stream>>>(
            dtbuf, xc, xdbl, A_log + (size_t)l * Di * 16, Dsk + (size_t)l * Di, xz, yb, S, Di);
        gemm(yb, Di, out_proj + (size_t)l * D * Di, Di, x, D, S, D, Di, 1);  // += residual

        // ---- LIF layer ----
        rms_kernel<<<S, 256, 0, stream>>>(x, lif_norm + (size_t)l * D, xn, D);
        gemm(xn, D, lif_in + (size_t)l * D * D, D, cur, D, S, D, D, 0);
        lif_scan_kernel<<<(D + 255) / 256, 256, 0, stream>>>(cur, spk, S, D);
        gemm(spk, D, lif_out + (size_t)l * D * D, D, x, D, S, D, D, 1);      // += residual

        // ---- MoE layer (dense over experts, per-expert GEMMs) ----
        rms_kernel<<<S, 256, 0, stream>>>(x, moe_norm + (size_t)l * D, xn, D);
        router_kernel<<<(T + 255) / 256, 256, 0, stream>>>(
            xn, router + (size_t)l * E * D, wts, T, D);
        for (int e = 0; e < E; ++e) {
            const float* gw = gate + ((size_t)l * E + e) * H * D;  // [H,D]
            const float* uw = up   + ((size_t)l * E + e) * H * D;  // [H,D]
            const float* dw = down + ((size_t)l * E + e) * D * H;  // [D,H]
            gemm(xn, D, gw, D, he, H, T, H, D, 0);
            gemm(xn, D, uw, D, hu, H, T, H, D, 0);
            moe_mul_kernel<<<(T * H + 255) / 256, 256, 0, stream>>>(he, hu, wts, e, T, H);
            gemm(he, H, dw, H, x, D, T, D, H, 1);                  // += into residual
        }
    }

    // ---- Final norm + LM head ----
    rms_kernel<<<S, 256, 0, stream>>>(x, norm_out, xn, D);
    gemm(xn, D, lm_head, D, outp, V, S, V, D, 0);
}
